// Codec_37967510897191
// MI455X (gfx1250) — matrix operand order repaired and checked by reading, not compiled, not run on hardware
//
#include <hip/hip_runtime.h>
#include <hip/hip_bf16.h>

// ---------------------------------------------------------------------------
// MI455X (gfx1250) implementation of the SoundStream-style codec.
// All 256x256 convolutions (dilated conv3, stride-2 down-conv k4, transposed
// up-conv k4) and the RVQ codebook-distance GEMM run on v_wmma_f32_16x16x32_bf16
// (bf16 operands, fp32 accumulate).  Wave32, 16 waves / 512-thread workgroups.
// Conv kernels are templated on tap count so the K reduction fully unrolls:
// 24 / 32 / 16 static WMMA issues per wave with batched b128 fragment loads.
// ---------------------------------------------------------------------------

typedef __attribute__((ext_vector_type(16))) __bf16 bf16x16;
typedef __attribute__((ext_vector_type(8)))  __bf16 bf16x8;
typedef __attribute__((ext_vector_type(8)))  float  f32x8;

union BF16Frag { bf16x16 v; bf16x8 h[2]; };

static __device__ __forceinline__
f32x8 wmma_bf16(bf16x8 a0, bf16x8 a1, bf16x8 b0, bf16x8 b1, f32x8 c) {
  BF16Frag a, b;
  a.h[0] = a0; a.h[1] = a1;
  b.h[0] = b0; b.h[1] = b1;
  // D = A(16x32 bf16) * B(32x16 bf16) + C(16x16 f32)
  return __builtin_amdgcn_wmma_f32_16x16x32_bf16(
      /*neg_a=*/false, a.v, /*neg_b=*/false, b.v,
      /*c_mod=*/(short)0, c, /*reuse_a=*/false, /*reuse_b=*/false);
}

// ---------------------------------------------------------------------------
// Weight repack: fp32 W[...] (arbitrary strides) -> bf16 wpk[co][k][ci] so the
// WMMA A-fragment is two contiguous 16B loads per lane.
// ---------------------------------------------------------------------------
__global__ void pack_w(const float* __restrict__ w, __bf16* __restrict__ wpk,
                       int nK, int sCo, int sCi, int sK) {
  int i = blockIdx.x * 256 + threadIdx.x;          // [co][k][ci] flat
  int total = 256 * nK * 256;
  if (i >= total) return;
  int ci = i & 255;
  int k  = (i >> 8) % nK;
  int co = i / (256 * nK);
  wpk[i] = (__bf16)w[(size_t)co * sCo + (size_t)ci * sCi + (size_t)k * sK];
}

// ---------------------------------------------------------------------------
// Generic implicit-GEMM conv via WMMA, templated on tap count NT.
//   NT=3: conv3 dilated:  strideN=1, tapOff={-d,0,d},   tapK={0,1,2}, nK=3
//   NT=4: down k4 s2 p1:  strideN=2, tapOff={-1,0,1,2}, tapK={0,1,2,3}, nK=4
//   NT=2: up   k4 s2 p1:  strideN=1, grid.z=2 parity; tapOff={0,-1}+p,
//                         tapK={1,3}-p, nK=4, outputs interleaved stride 2
//   out[b][co][tOut] = bias[co] + resScale*resid[...] + sum_{tap,ci} W*X
// ---------------------------------------------------------------------------
template <int NT>
__global__ __launch_bounds__(512)
void conv_wmma(const float* __restrict__ x, const __bf16* __restrict__ wpk,
               const float* __restrict__ bias, const float* __restrict__ resid,
               float resScale, float* __restrict__ out,
               int Lin, int Lout, int nK,
               int4 tapOff4, int4 tapK4, int strideN, int inMul, int outMul,
               int outStride, int doRelu) {
  __shared__ __bf16 ldsX[4 * 16 * 256];            // [tap][n][ci], <= 32 KB
  const int toff[4] = {tapOff4.x, tapOff4.y, tapOff4.z, tapOff4.w};
  const int tkk[4]  = {tapK4.x,  tapK4.y,  tapK4.z,  tapK4.w};

  const int tIdx   = blockIdx.x * 16;
  const int b      = blockIdx.y;
  const int parity = blockIdx.z;
  const int inBase = tIdx * inMul;
  const float* xb  = x + (size_t)b * 256 * Lin;

  // Stage only the tap rows we need (relu + zero-pad applied here), bf16.
  const int total = NT * 16 * 256;
#pragma unroll
  for (int idx = threadIdx.x; idx < total; idx += 512) {
    int ci  = idx & 255;
    int n   = (idx >> 8) & 15;
    int tap = idx >> 12;
    int t   = inBase + n * strideN + toff[tap] + parity;
    float v = 0.f;
    if (t >= 0 && t < Lin) v = xb[(size_t)ci * Lin + t];
    if (doRelu && v < 0.f) v = 0.f;
    ldsX[(tap * 16 + n) * 256 + ci] = (__bf16)v;
  }
  __syncthreads();

  const int wave = threadIdx.x >> 5;               // 16 waves: co tiles 0..15
  const int lane = threadIdx.x & 31;
  const int co0  = wave * 16;
  const int m    = lane & 15;                      // A row / B column
  const int g    = lane >> 4;                      // lane-half selector

  f32x8 acc = {};
  const __bf16* wrow = wpk + (size_t)(co0 + m) * nK * 256;
  __builtin_prefetch(wrow, 0, 3);                  // global_prefetch weight row
#pragma unroll
  for (int s = 0; s < NT * 8; ++s) {
    const int tap  = s >> 3;                       // compile-time per iteration
    const int cb   = (s & 7) << 5;
    const int kidx = tkk[tap] - parity;
    // A fragment: lane m, K halves {g*8..+7} then {16+g*8..+7}
    const __bf16* ap = wrow + kidx * 256 + cb + g * 8;
    bf16x8 a0 = *(const bf16x8*)(ap);
    bf16x8 a1 = *(const bf16x8*)(ap + 16);
    // B fragment: lane n=m, K halves {g*8..+7} then {16+g*8..+7}
    const __bf16* bp = &ldsX[(tap * 16 + m) * 256 + cb + g * 8];
    bf16x8 b0 = *(const bf16x8*)(bp);
    bf16x8 b1 = *(const bf16x8*)(bp + 16);
    acc = wmma_bf16(a0, a1, b0, b1, acc);
  }

  // Epilogue per C/D layout: VGPR v -> (co0 + g*8 + v, tIdx tile col lane&15)
  const int n    = lane & 15;
  const int tOut = tIdx * outMul + parity + n * outStride;
  float* ob = out + (size_t)b * 256 * Lout;
  if (resid != nullptr) {
    const float* rb = resid + (size_t)b * 256 * Lout;
#pragma unroll
    for (int v = 0; v < 8; ++v) {
      int co = co0 + g * 8 + v;
      size_t oi = (size_t)co * Lout + tOut;
      ob[oi] = acc[v] + bias[co] + resScale * rb[oi];
    }
  } else {
#pragma unroll
    for (int v = 0; v < 8; ++v) {
      int co = co0 + g * 8 + v;
      ob[(size_t)co * Lout + tOut] = acc[v] + bias[co];
    }
  }
}

// ---------------------------------------------------------------------------
// Scalar edge kernels (1-channel ends; negligible FLOPs)
// ---------------------------------------------------------------------------
__global__ void conv0_k4s2(const float* __restrict__ x, const float* __restrict__ w,
                           const float* __restrict__ bias, float* __restrict__ out,
                           int Lin, int Lout) {
  int i = blockIdx.x * 256 + threadIdx.x;          // 2*256*Lout
  if (i >= 2 * 256 * Lout) return;
  int t  = i % Lout;
  int co = (i / Lout) & 255;
  int b  = i / (256 * Lout);
  const float* xb = x + (size_t)b * Lin;
  float acc = bias[co];
#pragma unroll
  for (int k = 0; k < 4; ++k) {
    int ti = 2 * t + k - 1;
    if (ti >= 0 && ti < Lin) acc += w[co * 4 + k] * xb[ti];
  }
  out[i] = acc;
}

__global__ void up_last(const float* __restrict__ xin, const float* __restrict__ w,
                        const float* __restrict__ bias, float* __restrict__ recon,
                        int Lin) {                 // 256 -> 1, Lout = 2*Lin
  int Lout = 2 * Lin;
  int i = blockIdx.x * 256 + threadIdx.x;          // 2*Lout
  if (i >= 2 * Lout) return;
  int t = i % Lout;
  int b = i / Lout;
  const float* xb = xin + (size_t)b * 256 * Lin;
  int p  = t & 1;
  int k0 = p ? 0 : 1, k1 = p ? 2 : 3;
  int ta = p ? (t + 1) / 2 : t / 2;
  int tb = ta - 1;
  float acc = bias[0];
  for (int ci = 0; ci < 256; ++ci) {
    const float* wr = w + ci * 4;                  // w[ci][0][k]
    if (ta >= 0 && ta < Lin) acc += wr[k0] * xb[(size_t)ci * Lin + ta];
    if (tb >= 0 && tb < Lin) acc += wr[k1] * xb[(size_t)ci * Lin + tb];
  }
  recon[i] = acc;
}

// ---------------------------------------------------------------------------
// RVQ kernels
// ---------------------------------------------------------------------------
__global__ void build_emb(const float* __restrict__ z, float* __restrict__ emb,
                          float* __restrict__ q, float* __restrict__ commitAcc,
                          int Tz) {
  int i = blockIdx.x * 256 + threadIdx.x;          // (b*Tz+t)*256 + c
  if (i == 0) *commitAcc = 0.f;
  if (i >= 2 * Tz * 256) return;
  int c = i & 255;
  int t = (i >> 8) % Tz;
  int b = (i >> 8) / Tz;
  emb[i] = z[((size_t)b * 256 + c) * Tz + t];
  q[i] = 0.f;
}

__global__ void rvq_build_cb(const float* __restrict__ rvq_sum,
                             const float* __restrict__ rvq_usage,
                             __bf16* __restrict__ cbbf, float* __restrict__ cbnorm,
                             int it) {
  int k = blockIdx.x * 256 + threadIdx.x;
  if (k >= 1024) return;
  float u = fmaxf(rvq_usage[it * 1024 + k], 1e-5f);
  const float* row = rvq_sum + ((size_t)it * 1024 + k) * 256;
  float nrm = 0.f;
  for (int c = 0; c < 256; ++c) {
    float v = row[c] / u;
    nrm += v * v;
    cbbf[k * 256 + c] = (__bf16)v;
  }
  cbnorm[k] = nrm;
}

__global__ void rvq_res(const float* __restrict__ emb, const float* __restrict__ q,
                        float* __restrict__ res, __bf16* __restrict__ resbf) {
  int i = blockIdx.x * 256 + threadIdx.x;          // N*C = 1M
  float r = emb[i] - q[i];
  res[i] = r;
  resbf[i] = (__bf16)r;
}

// scores[k][n] = ||cb_k||^2 - 2 * <cb_k, res_n>  via WMMA (K-red = 256)
__global__ __launch_bounds__(512)
void rvq_scores(const __bf16* __restrict__ cbbf, const float* __restrict__ cbnorm,
                const __bf16* __restrict__ resbf, float* __restrict__ scores,
                int N) {
  int wave  = threadIdx.x >> 5;
  int lane  = threadIdx.x & 31;
  int kbase = blockIdx.y * 256 + wave * 16;
  int nbase = blockIdx.x * 16;
  int m = lane & 15;
  int g = lane >> 4;
  f32x8 acc = {};
  const __bf16* arow = cbbf  + (size_t)(kbase + m) * 256;
  const __bf16* brow = resbf + (size_t)(nbase + m) * 256;
#pragma unroll
  for (int s = 0; s < 8; ++s) {
    int cb = s << 5;
    bf16x8 a0 = *(const bf16x8*)(arow + cb + g * 8);
    bf16x8 a1 = *(const bf16x8*)(arow + cb + 16 + g * 8);
    bf16x8 b0 = *(const bf16x8*)(brow + cb + g * 8);
    bf16x8 b1 = *(const bf16x8*)(brow + cb + 16 + g * 8);
    acc = wmma_bf16(a0, a1, b0, b1, acc);
  }
  int n = nbase + m;
#pragma unroll
  for (int v = 0; v < 8; ++v) {
    int k = kbase + g * 8 + v;
    scores[(size_t)k * N + n] = cbnorm[k] - 2.f * acc[v];
  }
}

__global__ void rvq_argmin(const float* __restrict__ scores, int* __restrict__ codes,
                           int N, int it) {
  int n = blockIdx.x * 256 + threadIdx.x;
  if (n >= N) return;
  float best = scores[n];
  int bi = 0;
  for (int k = 1; k < 1024; ++k) {
    float s = scores[(size_t)k * N + n];
    if (s < best) { best = s; bi = k; }
  }
  codes[(size_t)n * 8 + it] = bi;
}

__global__ __launch_bounds__(256)
void rvq_update(const float* __restrict__ rvq_sum, const float* __restrict__ rvq_usage,
                const int* __restrict__ codes, const float* __restrict__ res,
                float* __restrict__ q, float* __restrict__ commitAcc, int it) {
  int idx = blockIdx.x * 256 + threadIdx.x;        // N*C
  int c = idx & 255;
  int n = idx >> 8;
  int k = codes[(size_t)n * 8 + it];
  float u   = fmaxf(rvq_usage[it * 1024 + k], 1e-5f);
  float cbv = rvq_sum[((size_t)it * 1024 + k) * 256 + c] / u;
  q[idx] += cbv;
  float d = cbv - res[idx];
  __shared__ float red[256];
  red[threadIdx.x] = d * d;
  __syncthreads();
  for (int o = 128; o > 0; o >>= 1) {
    if (threadIdx.x < o) red[threadIdx.x] += red[threadIdx.x + o];
    __syncthreads();
  }
  if (threadIdx.x == 0)
    atomicAdd(commitAcc, red[0] * (1.0f / (4096.0f * 256.0f)));
}

__global__ void build_zq(const float* __restrict__ q, float* __restrict__ zq, int Tz) {
  int i = blockIdx.x * 256 + threadIdx.x;          // [b][c][t]
  if (i >= 2 * 256 * Tz) return;
  int t = i % Tz;
  int c = (i / Tz) & 255;
  int b = i / (256 * Tz);
  zq[i] = q[((size_t)b * Tz + t) * 256 + c];
}

__global__ void write_commit(const float* __restrict__ acc, float* __restrict__ dst) {
  if (threadIdx.x == 0 && blockIdx.x == 0) dst[0] = acc[0];
}

// ---------------------------------------------------------------------------
// Host orchestration
// ---------------------------------------------------------------------------
extern "C" void kernel_launch(void* const* d_in, const int* in_sizes, int n_in,
                              void* d_out, int out_size, void* d_ws, size_t ws_size,
                              hipStream_t stream) {
  (void)in_sizes; (void)n_in; (void)out_size; (void)ws_size;

  const float* x    = (const float*)d_in[0];
  const float* e0w  = (const float*)d_in[1];
  const float* e0b  = (const float*)d_in[2];
  const float* edw  = (const float*)d_in[3];   // [4][256][256][4]
  const float* edb  = (const float*)d_in[4];   // [4][256]
  const float* erw  = (const float*)d_in[5];   // [5][4][2][256][256][3]
  const float* erb  = (const float*)d_in[6];   // [5][4][2][256]
  const float* drw  = (const float*)d_in[7];
  const float* drb  = (const float*)d_in[8];
  const float* duw  = (const float*)d_in[9];   // [4][256][256][4]  (in,out,k)
  const float* dub  = (const float*)d_in[10];  // [4][256]
  const float* dulw = (const float*)d_in[11];  // [256][1][4]
  const float* dulb = (const float*)d_in[12];  // [1]
  const float* rsum = (const float*)d_in[13];  // [8][1024][256]
  const float* rusg = (const float*)d_in[14];  // [8][1024]

  // Workspace layout (~234 MB)
  char* w8 = (char*)d_ws;
  float*  A0     = (float*)(w8);
  float*  A1     = (float*)(w8 + (size_t)67108864);
  float*  A2     = (float*)(w8 + (size_t)134217728);
  __bf16* WPK    = (__bf16*)(w8 + (size_t)201326592);  // 512 KB
  __bf16* CBBF   = (__bf16*)(w8 + (size_t)201850880);  // 512 KB
  float*  CBNORM = (float*)(w8 + (size_t)202375168);   // 4 KB
  float*  EMB    = (float*)(w8 + (size_t)202379264);   // 4 MB
  float*  Q      = (float*)(w8 + (size_t)206573568);   // 4 MB
  float*  RES    = (float*)(w8 + (size_t)210767872);   // 4 MB
  __bf16* RESBF  = (__bf16*)(w8 + (size_t)214962176);  // 2 MB
  float*  SCORES = (float*)(w8 + (size_t)217059328);   // 16 MB
  float*  COMMIT = (float*)(w8 + (size_t)233836544);

  float* outF      = (float*)d_out;
  int*   codesOut  = (int*)(outF + 131072);            // [4096][8]
  float* commitOut = outF + 131072 + 32768;

  auto pack = [&](const float* w, int nK, int sCo, int sCi, int sK) {
    int tot = 256 * nK * 256;
    pack_w<<<dim3((tot + 255) / 256), dim3(256), 0, stream>>>(w, WPK, nK, sCo, sCi, sK);
  };

  // dilated conv3 (256->256), relu on input, optional residual add
  auto conv3 = [&](const float* in, const float* w, const float* b,
                   const float* resid, float scale, float* out, int L, int dil) {
    pack(w, 3, /*sCo=*/768, /*sCi=*/3, /*sK=*/1);
    conv_wmma<3><<<dim3(L / 16, 2, 1), dim3(512), 0, stream>>>(
        in, WPK, b, resid, scale, out, L, L, 3,
        make_int4(-dil, 0, dil, 0), make_int4(0, 1, 2, 0),
        /*strideN=*/1, /*inMul=*/1, /*outMul=*/1, /*outStride=*/1, /*relu=*/1);
  };

  // stride-2 down conv k=4 (256->256), no relu, Lout = Lin/2
  auto conv_down = [&](const float* in, const float* w, const float* b,
                       float* out, int Lout) {
    pack(w, 4, /*sCo=*/1024, /*sCi=*/4, /*sK=*/1);
    conv_wmma<4><<<dim3(Lout / 16, 2, 1), dim3(512), 0, stream>>>(
        in, WPK, b, nullptr, 0.f, out, 2 * Lout, Lout, 4,
        make_int4(-1, 0, 1, 2), make_int4(0, 1, 2, 3),
        /*strideN=*/2, /*inMul=*/2, /*outMul=*/1, /*outStride=*/1, /*relu=*/0);
  };

  // transposed conv k=4 s=2 p=1 (256->256), per-parity, Lout = 2*Lin
  auto conv_up = [&](const float* in, const float* w, const float* b,
                     float* out, int Lin) {
    pack(w, 4, /*sCo=*/4, /*sCi=*/1024, /*sK=*/1);   // w[ci][co][k] -> [co][k][ci]
    conv_wmma<2><<<dim3(Lin / 16, 2, 2), dim3(512), 0, stream>>>(
        in, WPK, b, nullptr, 0.f, out, Lin, 2 * Lin, 4,
        make_int4(0, -1, 0, 0), make_int4(1, 3, 0, 0),
        /*strideN=*/1, /*inMul=*/1, /*outMul=*/2, /*outStride=*/2, /*relu=*/0);
  };

  float* cur = A0; float* s1 = A1; float* s2 = A2;

  auto resblock = [&](const float* rw2, const float* rb2, int L, int dil, float scale) {
    // h = conv1(relu(x)) ; out = scale*x + conv2(relu(h))
    conv3(cur, rw2, rb2, nullptr, 0.f, s1, L, dil);
    conv3(s1, rw2 + 256 * 256 * 3, rb2 + 256, cur, scale, s2, L, /*dil=*/1);
    float* t = cur; cur = s2; s2 = t;
  };

  // ----------------------- encoder -----------------------
  int L = 32768;
  conv0_k4s2<<<dim3((2 * 256 * L + 255) / 256), dim3(256), 0, stream>>>(
      x, e0w, e0b, cur, 65536, L);
  for (int i = 0; i < 5; ++i) {
    if (i > 0) {
      conv_down(cur, edw + (size_t)(i - 1) * 256 * 256 * 4, edb + (i - 1) * 256, s1, L / 2);
      float* t = cur; cur = s1; s1 = t;
      L /= 2;
    }
    for (int j = 0; j < 4; ++j) {
      int dil = 1; for (int p = 0; p <= j; ++p) dil *= 3;  // 3^(j+1)
      size_t base = ((size_t)(i * 4 + j) * 2);
      resblock(erw + base * 256 * 256 * 3, erb + base * 256, L, dil, 1.0f);
    }
  }
  // cur holds z: [2][256][2048]
  const int Tz = 2048;

  // ----------------------- RVQ -----------------------
  build_emb<<<dim3((2 * Tz * 256 + 255) / 256), dim3(256), 0, stream>>>(
      cur, EMB, Q, COMMIT, Tz);
  for (int it = 0; it < 8; ++it) {
    rvq_build_cb<<<dim3(4), dim3(256), 0, stream>>>(rsum, rusg, CBBF, CBNORM, it);
    rvq_res<<<dim3(4096), dim3(256), 0, stream>>>(EMB, Q, RES, RESBF);
    rvq_scores<<<dim3(4096 / 16, 4, 1), dim3(512), 0, stream>>>(
        CBBF, CBNORM, RESBF, SCORES, 4096);
    rvq_argmin<<<dim3(16), dim3(256), 0, stream>>>(SCORES, codesOut, 4096, it);
    rvq_update<<<dim3(4096), dim3(256), 0, stream>>>(
        rsum, rusg, codesOut, RES, Q, COMMIT, it);
  }
  write_commit<<<dim3(1), dim3(1), 0, stream>>>(COMMIT, commitOut);
  build_zq<<<dim3((2 * 256 * Tz + 255) / 256), dim3(256), 0, stream>>>(Q, s1, Tz);
  { float* t = cur; cur = s1; s1 = t; }

  // ----------------------- decoder -----------------------
  L = Tz;
  for (int i = 0; i < 5; ++i) {
    for (int j = 0; j < 4; ++j) {
      int dil = 1; for (int p = 0; p <= j; ++p) dil *= 3;
      size_t base = ((size_t)(i * 4 + j) * 2);
      resblock(drw + base * 256 * 256 * 3, drb + base * 256, L, dil, 2.0f);
    }
    if (i < 4) {
      conv_up(cur, duw + (size_t)i * 256 * 256 * 4, dub + i * 256, s1, L);
      float* t = cur; cur = s1; s1 = t;
      L *= 2;
    } else {
      up_last<<<dim3((2 * 2 * L + 255) / 256), dim3(256), 0, stream>>>(
          cur, dulw, dulb, outF, L);
    }
  }
}
